// LMAEmbedding_90254442758929
// MI455X (gfx1250) — compile-verified, matrix operand order checked by reading, not run on hardware
//
#include <hip/hip_runtime.h>

typedef __attribute__((ext_vector_type(16))) _Float16 v16h;
typedef __attribute__((ext_vector_type(8)))  float    v8f;

#define INPUT_DIM   128
#define EMBED_DIM   64
#define CHUNK       8
#define BITS        8
#define NREP        2
#define NCHUNKS     8
#define ARRAY_SZ    262144
#define NPROJ       128        // NREP * NCHUNKS * BITS
#define ROWS_PB     16         // batch rows per block

__global__ __launch_bounds__(256)
void LMAEmbedding_gfx1250_kernel(const float* __restrict__ x,
                                 const float* __restrict__ hw,
                                 const float* __restrict__ lsh,
                                 const long long* __restrict__ rn,
                                 long long* __restrict__ idx_out,
                                 float* __restrict__ out)
{
    // B-matrix staged fragment-ready: lshB[ks][half][n] is the exact 16-half
    // (32 B) fragment lane (half*16 + n%16) consumes at K-step ks.
    __shared__ v16h     lshB[4 * 2 * NPROJ];              // 32 KB
    __shared__ _Float16 xs[ROWS_PB * INPUT_DIM];          //  4 KB, [m][k] f16
    __shared__ float    proj_s[ROWS_PB][NPROJ];           //  8 KB
    __shared__ float    gath_s[ROWS_PB][NREP][EMBED_DIM]; //  8 KB

    const int    tid     = threadIdx.x;
    const size_t rowBase = (size_t)blockIdx.x * ROWS_PB;

    // ---- Stage lsh (f32 -> f16) into fragment-ready layout, coalesced reads ----
    {
        _Float16* lshB_h = (_Float16*)lshB;
        for (int s = tid; s < INPUT_DIM * NPROJ; s += 256) {
            const int k  = s >> 7;            // input-dim index 0..127
            const int n  = s & 127;           // projection column 0..127
            const int ks = k >> 5;            // K-step 0..3
            const int kk = k & 31;            // K within step
            // CDNA5 16-bit fragment layout: lane-half 0 holds K {0..7,16..23},
            // lane-half 1 holds K {8..15,24..31}; half j<8 -> K kb+j, j>=8 -> K kb+8+j
            const int half = (kk >> 3) & 1;
            const int j    = (kk & 7) | ((kk >> 4) << 3);
            lshB_h[((((ks << 1) | half) << 7) + n) * 16 + j] = (_Float16)lsh[s];
        }
    }
    // ---- Stage x tile (f32 -> f16) ----
    for (int i = tid; i < ROWS_PB * INPUT_DIM; i += 256) {
        const int m = i >> 7, k = i & 127;
        xs[i] = (_Float16)x[(rowBase + m) * INPUT_DIM + k];
    }
    __syncthreads();

    // ---- Phase 1: proj = xs * lshB via WMMA. Wave w owns N-tile [16w,16w+16) ----
    const int wave    = tid >> 5;
    const int lane    = tid & 31;
    const int half16  = lane & 15;
    const int halfsel = lane >> 4;        // 0: K base 0, 1: K base 8
    const int kgrp    = halfsel * 8;
    const int ntile   = wave * 16;
    const int n       = ntile + half16;

    v8f acc = {};
    #pragma unroll
    for (int ks = 0; ks < 4; ++ks) {      // K = 128 in 4 steps of 32
        const int kb = ks * 32 + kgrp;
        v16h a;
        const _Float16* arow = &xs[half16 * INPUT_DIM];
        #pragma unroll
        for (int j = 0; j < 8; ++j) {     // two contiguous 16 B groups -> ds_load_b128 x2
            a[j]     = arow[kb + j];
            a[j + 8] = arow[kb + 16 + j];
        }
        const v16h b = lshB[(((ks << 1) | halfsel) << 7) + n];  // one aligned 32 B load
        acc = __builtin_amdgcn_wmma_f32_16x16x32_f16(
            /*neg_a=*/false, a, /*neg_b=*/false, b,
            /*c_mod=*/(short)0, acc, /*reuse_a=*/false, /*reuse_b=*/false);
    }

    // C tile layout: lanes 0-15 -> M = v, N = lane; lanes 16-31 -> M = v+8, N = lane-16
    #pragma unroll
    for (int v = 0; v < 8; ++v) {
        const int m = (lane < 16) ? v : (v + 8);
        proj_s[m][n] = acc[v];
    }
    __syncthreads();

    // ---- Phase 2: SRP bits -> universal hash -> indices + gather ----
    const long long P  = rn[0];
    const long long ha = rn[1];
    const long long hb = rn[2];
    const long long hc = rn[3];
    {
        const int m    = tid >> 4;   // row 0..15
        const int pair = tid & 15;
        const int r    = pair >> 3;  // rep 0..1
        const int ch   = pair & 7;   // chunk 0..7
        const float* p = &proj_s[m][r * (NCHUNKS * BITS) + ch * BITS];

        long long srp = 0;
        #pragma unroll
        for (int bit = 0; bit < BITS; ++bit)
            srp |= (long long)(p[bit] > 0.0f) << bit;

        const long long loc  = ((ha * srp + hb * (long long)ch + hc) % P)
                               % (long long)(ARRAY_SZ - CHUNK + 1);
        const long long base = (long long)r * ARRAY_SZ + loc;
        const size_t    o    = ((rowBase + m) * NREP + r) * EMBED_DIM + (size_t)ch * CHUNK;

        #pragma unroll
        for (int j = 0; j < CHUNK; ++j) {
            idx_out[o + j] = base + j;                       // int64 index output
            gath_s[m][r][ch * CHUNK + j] = hw[base + j];     // L2-resident gather
        }
    }
    __syncthreads();

    // ---- Phase 3: mean over reps, store float output ----
    for (int i = tid; i < ROWS_PB * EMBED_DIM; i += 256) {
        const int m = i >> 6, e = i & 63;
        out[(rowBase + m) * EMBED_DIM + e] = 0.5f * (gath_s[m][0][e] + gath_s[m][1][e]);
    }
}

extern "C" void kernel_launch(void* const* d_in, const int* in_sizes, int n_in,
                              void* d_out, int out_size, void* d_ws, size_t ws_size,
                              hipStream_t stream) {
    const float*     x   = (const float*)d_in[0];      // [B, 128] f32
    const float*     hw  = (const float*)d_in[1];      // [524288] f32
    const float*     lsh = (const float*)d_in[2];      // [128, 2, 64] f32
    const long long* rn  = (const long long*)d_in[3];  // [51] int64

    const size_t batch = (size_t)in_sizes[0] / INPUT_DIM;

    // d_out = concat( hashed_idx [B,2,64] int64 , output [B,64] f32 )
    long long* idx_out = (long long*)d_out;
    float*     out     = (float*)((char*)d_out +
                          batch * (size_t)NREP * EMBED_DIM * sizeof(long long));

    const int blocks = (int)(batch / ROWS_PB);
    LMAEmbedding_gfx1250_kernel<<<blocks, 256, 0, stream>>>(x, hw, lsh, rn, idx_out, out);
}
